// AttnLinearEncoder_22127671509756
// MI455X (gfx1250) — matrix operand order repaired and checked by reading, not compiled, not run on hardware
//
#include <hip/hip_runtime.h>
#include <math.h>

#define N_ROWS 12288
#define K_DIM  512
#define F_DIM  128

typedef float    v2f  __attribute__((ext_vector_type(2)));
typedef float    v8f  __attribute__((ext_vector_type(8)));
typedef _Float16 v8h  __attribute__((ext_vector_type(8)));
typedef _Float16 v16h __attribute__((ext_vector_type(16)));

static __device__ __forceinline__ v8f wmma_f32(v2f a, v2f b, v8f c) {
  // D = A(16x4) * B(4x16) + C(16x16), all f32, wave32
  return __builtin_amdgcn_wmma_f32_16x16x4_f32(false, a, false, b, (short)0, c,
                                               false, false);
}

static __device__ __forceinline__ v8f wmma_f16(v16h a, v16h b, v8f c) {
  // D = A(16x32 f16) * B(32x16 f16) + C(16x16 f32)
  return __builtin_amdgcn_wmma_f32_16x16x32_f16(false, a, false, b, (short)0, c,
                                                false, false);
}

// ---------------------------------------------------------------------------
// Kernel 1: wT[k][n] = g[n] * v[n][k] / ||v[n]||   (k-major for coalesced B)
// ---------------------------------------------------------------------------
__global__ __launch_bounds__(256) void make_w_kernel(const float* __restrict__ v,
                                                     const float* __restrict__ g,
                                                     float* __restrict__ wT) {
  __shared__ float red[256];
  __shared__ float scale_sh;
  const int row = blockIdx.x;   // 0..127
  const int tid = threadIdx.x;
  float ss = 0.f;
  for (int c = tid; c < K_DIM; c += 256) {
    float t = v[row * K_DIM + c];
    ss += t * t;
  }
  red[tid] = ss;
  __syncthreads();
  for (int o = 128; o > 0; o >>= 1) {
    if (tid < o) red[tid] += red[tid + o];
    __syncthreads();
  }
  if (tid == 0) scale_sh = g[row] / sqrtf(red[0]);
  __syncthreads();
  const float sc = scale_sh;
  for (int c = tid; c < K_DIM; c += 256)
    wT[c * F_DIM + row] = v[row * K_DIM + c] * sc;
}

// ---------------------------------------------------------------------------
// Kernel 2: z = x @ w.T + b  (f32 WMMA, full precision since z feeds exp),
// plus z16T = f16 transpose of z ([128][12288]) for the attention B operands.
// One block = 16 rows x 128 cols; 8 waves each own one 16-col tile.
// ---------------------------------------------------------------------------
__global__ __launch_bounds__(256) void zgemm_kernel(const float* __restrict__ x,
                                                    const float* __restrict__ wT,
                                                    const float* __restrict__ bias,
                                                    float* __restrict__ z,
                                                    _Float16* __restrict__ z16T) {
  __shared__ float xs[16 * 516];  // pad 512 -> 516 (bank-conflict-free column reads)
  const int tid = threadIdx.x;
  const int m0 = blockIdx.x * 16;

  // Cooperative coalesced float4 staging of the 16x512 x-tile
  for (int i4 = tid; i4 < 16 * (K_DIM / 4); i4 += 256) {
    const int row = i4 >> 7;     // /128
    const int c4 = i4 & 127;
    float4 val = ((const float4*)(x + (size_t)(m0 + row) * K_DIM))[c4];
    ((float4*)(xs + row * 516))[c4] = val;
  }
  __syncthreads();

  const int lane = tid & 31;
  const int wv = tid >> 5;
  const int n0 = wv * 16;
  const int mloc = lane & 15;
  const int khalf = (lane >> 4) * 2;  // ISA A/B layout: K = (lane/16)*2 + vgpr

  v8f c = {0.f, 0.f, 0.f, 0.f, 0.f, 0.f, 0.f, 0.f};
  const float* xrow = xs + mloc * 516;
  for (int k0 = 0; k0 < K_DIM; k0 += 4) {
    v2f a, b;
    a.x = xrow[k0 + khalf];
    a.y = xrow[k0 + khalf + 1];
    b.x = wT[(k0 + khalf) * F_DIM + n0 + mloc];
    b.y = wT[(k0 + khalf + 1) * F_DIM + n0 + mloc];
    c = wmma_f32(a, b, c);
  }

  const float bn = bias[n0 + mloc];
  const int col = n0 + mloc;
  const int rowbase = m0 + (lane >> 4) * 8;  // C/D layout: M = r + (lane/16)*8
  v8h h;
#pragma unroll
  for (int r = 0; r < 8; r++) {
    const float val = c[r] + bn;
    z[(size_t)(rowbase + r) * F_DIM + col] = val;
    h[r] = (_Float16)val;
  }
  // 8 consecutive rows of one column of z  ==  16B-aligned contiguous run in z16T
  *(v8h*)(z16T + (size_t)col * N_ROWS + rowbase) = h;
}

// ---------------------------------------------------------------------------
// Kernel 3: s = z @ a_src, d = z @ a_dst    (one wave per row)
// ---------------------------------------------------------------------------
__global__ __launch_bounds__(256) void sd_kernel(const float* __restrict__ z,
                                                 const float* __restrict__ a_src,
                                                 const float* __restrict__ a_dst,
                                                 float* __restrict__ s,
                                                 float* __restrict__ d) {
  const int row = blockIdx.x * 8 + (threadIdx.x >> 5);
  const int lane = threadIdx.x & 31;
  float ps = 0.f, pd = 0.f;
#pragma unroll
  for (int c = 0; c < F_DIM; c += 32) {
    float zv = z[(size_t)row * F_DIM + c + lane];
    ps += zv * a_src[c + lane];
    pd += zv * a_dst[c + lane];
  }
#pragma unroll
  for (int o = 16; o > 0; o >>= 1) {
    ps += __shfl_xor(ps, o, 32);
    pd += __shfl_xor(pd, o, 32);
  }
  if (lane == 0) {
    s[row] = ps;
    d[row] = pd;
  }
}

// ---------------------------------------------------------------------------
// Kernel 4: d_max = max_j d[j]
// ---------------------------------------------------------------------------
__global__ __launch_bounds__(1024) void dmax_kernel(const float* __restrict__ d,
                                                    float* __restrict__ dmax) {
  __shared__ float red[1024];
  const int tid = threadIdx.x;
  float m = -3.402823466e38f;
  for (int i = tid; i < N_ROWS; i += 1024) m = fmaxf(m, d[i]);
  red[tid] = m;
  __syncthreads();
  for (int o = 512; o > 0; o >>= 1) {
    if (tid < o) red[tid] = fmaxf(red[tid], red[tid + o]);
    __syncthreads();
  }
  if (tid == 0) dmax[0] = red[0];
}

// ---------------------------------------------------------------------------
// Kernel 5: flash-style  z2 = softmax_rows(relu(s_i+d_j)) @ z + z,
//           out = softmax(z2, axis=-1).
// Block = 32 rows (2 M-tiles) x 128 cols (8 N-tiles). 8 waves partition j in
// chunks of 32 (one f16 WMMA K-block). m_i = max(0, s_i + d_max) is known a
// priori -> no online rescale. P computed in f32, rounded to f16 for both the
// numerator (WMMA A operand) and the denominator so they stay consistent.
// ---------------------------------------------------------------------------
__global__ __launch_bounds__(256) void attn_kernel(const float* __restrict__ z,
                                                   const _Float16* __restrict__ z16T,
                                                   const float* __restrict__ s,
                                                   const float* __restrict__ d,
                                                   const float* __restrict__ dmax_p,
                                                   float* __restrict__ out) {
  __shared__ float srow[32], mrow[32];
  __shared__ float red[8 * 256];    // 8 KB: cross-wave C reduction (one 16x16 tile)
  __shared__ float z2s[32 * 128];   // 16 KB
  __shared__ float dred[8 * 32];
  __shared__ float denom[32];

  const int tid = threadIdx.x;
  const int lane = tid & 31;
  const int wv = tid >> 5;
  const int m0 = blockIdx.x * 32;

  const float dmaxv = dmax_p[0];
  if (tid < 32) {
    float sv = s[m0 + tid];
    srow[tid] = sv;
    mrow[tid] = fmaxf(sv + dmaxv, 0.f);
  }
  __syncthreads();

  const int mloc = lane & 15;
  const int half = lane >> 4;
  const float s_m0 = srow[mloc],      m_m0 = mrow[mloc];
  const float s_m1 = srow[16 + mloc], m_m1 = mrow[16 + mloc];

  v8f acc[2][8];
#pragma unroll
  for (int mt = 0; mt < 2; mt++)
#pragma unroll
    for (int t = 0; t < 8; t++)
      acc[mt][t] = (v8f){0.f, 0.f, 0.f, 0.f, 0.f, 0.f, 0.f, 0.f};

  float dsum0 = 0.f, dsum1 = 0.f;

  // 12288 / (8 waves * 32) = 48 uniform iterations per wave
  for (int it = 0; it < N_ROWS / 256; ++it) {
    const int jblk = it * 256 + wv * 32;

    // ---- A operands: lane row = mloc;
    //      element (j8,e): K = 16*half + 2*j8 + e (matches contiguous B)
    v16h a0, a1;
#pragma unroll
    for (int j8 = 0; j8 < 8; ++j8) {
#pragma unroll
      for (int e = 0; e < 2; ++e) {
        const int K = 16 * half + 2 * j8 + e;
        const float dj = d[jblk + K];
        const _Float16 h0 = (_Float16)__expf(fmaxf(s_m0 + dj, 0.f) - m_m0);
        const _Float16 h1 = (_Float16)__expf(fmaxf(s_m1 + dj, 0.f) - m_m1);
        a0[j8 * 2 + e] = h0;
        a1[j8 * 2 + e] = h1;
        dsum0 += (float)h0;
        dsum1 += (float)h1;
      }
    }

    // ---- B operands: z16T[n][jblk + 16*half + 0..15] = 32B aligned run
    const _Float16* zt = z16T + jblk + half * 16;
#pragma unroll
    for (int t = 0; t < 8; ++t) {
      v16h b = *(const v16h*)(zt + (size_t)(t * 16 + mloc) * N_ROWS);
      acc[0][t] = wmma_f16(a0, b, acc[0][t]);
      acc[1][t] = wmma_f16(a1, b, acc[1][t]);
    }
  }

  // ---- denominator: lanes m and m+16 hold row-m partials -> shfl, then LDS
  dsum0 += __shfl_xor(dsum0, 16, 32);
  dsum1 += __shfl_xor(dsum1, 16, 32);
  if (lane < 16) {
    dred[wv * 32 + lane] = dsum0;
    dred[wv * 32 + 16 + lane] = dsum1;
  }
  __syncthreads();
  if (tid < 32) {
    float t = 0.f;
    for (int w = 0; w < 8; w++) t += dred[w * 32 + tid];
    denom[tid] = t;
  }

  // ---- fixed-order cross-wave reduction of accumulators, tile by tile
  const int rbase = half * 8;  // C/D layout: M = r + (lane/16)*8
  for (int mt = 0; mt < 2; mt++) {
    for (int t = 0; t < 8; t++) {
      __syncthreads();
#pragma unroll
      for (int r = 0; r < 8; r++)
        red[wv * 256 + (rbase + r) * 16 + mloc] = acc[mt][t][r];
      __syncthreads();
      const int er = tid >> 4, ec = tid & 15;
      float sum = 0.f;
      for (int w = 0; w < 8; w++) sum += red[w * 256 + er * 16 + ec];
      z2s[(mt * 16 + er) * F_DIM + t * 16 + ec] = sum;
    }
  }
  __syncthreads();

  // ---- z2 = z + acc/denom
  for (int idx = tid; idx < 32 * F_DIM; idx += 256) {
    const int mm = idx >> 7;
    z2s[idx] = z[(size_t)(m0 + mm) * F_DIM + (idx & 127)] + z2s[idx] / denom[mm];
  }
  __syncthreads();

  // ---- row softmax over 128, one wave handles 4 rows
  for (int rr = 0; rr < 4; rr++) {
    const int mm = wv * 4 + rr;
    const float* zr = z2s + mm * F_DIM;
    float v0 = zr[lane], v1 = zr[32 + lane], v2 = zr[64 + lane], v3 = zr[96 + lane];
    float mx = fmaxf(fmaxf(v0, v1), fmaxf(v2, v3));
#pragma unroll
    for (int o = 16; o > 0; o >>= 1) mx = fmaxf(mx, __shfl_xor(mx, o, 32));
    float e0 = __expf(v0 - mx), e1 = __expf(v1 - mx);
    float e2 = __expf(v2 - mx), e3 = __expf(v3 - mx);
    float ssum = e0 + e1 + e2 + e3;
#pragma unroll
    for (int o = 16; o > 0; o >>= 1) ssum += __shfl_xor(ssum, o, 32);
    const float inv = 1.f / ssum;
    float* ob = out + (size_t)(m0 + mm) * F_DIM;
    ob[lane] = e0 * inv;
    ob[32 + lane] = e1 * inv;
    ob[64 + lane] = e2 * inv;
    ob[96 + lane] = e3 * inv;
  }
}

// ---------------------------------------------------------------------------
extern "C" void kernel_launch(void* const* d_in, const int* in_sizes, int n_in,
                              void* d_out, int out_size, void* d_ws, size_t ws_size,
                              hipStream_t stream) {
  const float* x = (const float*)d_in[0];      // (12288, 512)
  const float* v = (const float*)d_in[1];      // (128, 512)
  const float* g = (const float*)d_in[2];      // (128, 1)
  const float* b = (const float*)d_in[3];      // (128,)
  const float* att = (const float*)d_in[4];    // (256, 1)

  char* ws = (char*)d_ws;
  size_t off = 0;
  float* wT = (float*)(ws + off);            off += (size_t)K_DIM * F_DIM * 4;
  float* z = (float*)(ws + off);             off += (size_t)N_ROWS * F_DIM * 4;
  _Float16* z16T = (_Float16*)(ws + off);    off += (size_t)N_ROWS * F_DIM * 2;
  float* s = (float*)(ws + off);             off += (size_t)N_ROWS * 4;
  float* d = (float*)(ws + off);             off += (size_t)N_ROWS * 4;
  float* dmax = (float*)(ws + off);
  float* out = (float*)d_out;

  make_w_kernel<<<F_DIM, 256, 0, stream>>>(v, g, wT);
  zgemm_kernel<<<N_ROWS / 16, 256, 0, stream>>>(x, wT, b, z, z16T);
  sd_kernel<<<N_ROWS / 8, 256, 0, stream>>>(z, att, att + F_DIM, s, d);
  dmax_kernel<<<1, 1024, 0, stream>>>(d, dmax);
  attn_kernel<<<N_ROWS / 32, 256, 0, stream>>>(z, z16T, s, d, dmax, out);
}
